// ClassicalGNN_90280212562547
// MI455X (gfx1250) — matrix operand order repaired and checked by reading, not compiled, not run on hardware
//
#include <hip/hip_runtime.h>
#include <hip/hip_bf16.h>
#include <math.h>

typedef __attribute__((ext_vector_type(16))) _Float16 v16h;
typedef __attribute__((ext_vector_type(8)))  float    v8f;

#define GNN_N     50000
#define GNN_E     800000
#define GNN_EP    (GNN_E + GNN_N)     // edges + self loops
#define GNN_IND   128
#define GNN_HID   128
#define GNN_OUTD  64
#define GNN_HEADS 4
#define GNN_FH    32
#define NEG_SLOPE 0.2f

// ---------------------------------------------------------------------------
// WMMA GEMM:  C[M,NCOLS] = act( A[M,128] @ B[128,NCOLS] + bias )
// fp32 in memory, f16 WMMA operands, f32 accumulation.
// Whole weight matrix B is transposed+converted once per block into LDS
// (Bt[n][k], f16, 320KB LDS per WGP makes this trivial); each wave then
// computes one 16-row x NCOLS strip: A fragment loaded once per K-step,
// 8 (or 4) WMMAs against LDS-resident B fragments (2x ds_load_b128 each).
// ---------------------------------------------------------------------------
template <int NCOLS>
__global__ __launch_bounds__(256)
void gat_gemm_wmma(const float* __restrict__ A,
                   const float* __restrict__ B,
                   const float* __restrict__ bias,
                   float* __restrict__ C,
                   int M, int relu_act)
{
    constexpr int K  = 128;
    constexpr int NT = NCOLS / 16;

    __shared__ __align__(32) _Float16 Bt[NCOLS * K];   // Bt[n*K + k] = B[k][n]

    // cooperative transpose + f32->f16 convert of the weight matrix
    for (int idx = threadIdx.x; idx < NCOLS * K; idx += blockDim.x) {
        const int n = idx / K;
        const int k = idx - n * K;
        Bt[idx] = (_Float16)B[(size_t)k * NCOLS + n];
    }
    __syncthreads();

    const int wave = blockIdx.x * (blockDim.x >> 5) + (threadIdx.x >> 5);
    const int lane = threadIdx.x & 31;
    const int nStrips = M >> 4;
    if (wave >= nStrips) return;          // wave-uniform: EXEC stays all-1s

    const int  m0    = wave << 4;
    const int  lhalf = lane & 15;
    const bool hi    = lane >= 16;

    const v8f zero = {};
    v8f acc[NT];
#pragma unroll
    for (int t = 0; t < NT; ++t) acc[t] = zero;

    // A row handled by this lane (ISA 16-bit A 16x32 layout)
    const float* __restrict__ Arow = A + (size_t)(m0 + lhalf) * K;

#pragma unroll
    for (int k0 = 0; k0 < K; k0 += 32) {
        // A fragment: lanes<16 -> K k0+0..15; lanes>=16 -> K k0+16..31
        const int ka = k0 + (hi ? 16 : 0);
        v16h a;
        const float* ap = Arow + ka;
#pragma unroll
        for (int i = 0; i < 8; ++i) a[i]     = (_Float16)ap[i];
#pragma unroll
        for (int i = 0; i < 8; ++i) a[8 + i] = (_Float16)ap[8 + i];

        // B fragments from LDS: lane L<16 holds column n0+L, K=k0+0..15;
        // lanes>=16 hold K=k0+16..31 (contiguous 32B in Bt => 2x ds_load_b128)
        const int kb = k0 + (hi ? 16 : 0);
#pragma unroll
        for (int t = 0; t < NT; ++t) {
            const v16h b =
                *(const v16h*)&Bt[(size_t)(t * 16 + lhalf) * K + kb];
            acc[t] = __builtin_amdgcn_wmma_f32_16x16x32_f16(
                         false, a, false, b, (short)0, acc[t], false, false);
        }
    }

    // write back: C VGPR r -> row m0 + r (+8 if lane>=16), col = tile*16+lhalf
    const int rbase = m0 + (hi ? 8 : 0);
#pragma unroll
    for (int t = 0; t < NT; ++t) {
        const int ccol = t * 16 + lhalf;
        const float bval = bias ? bias[ccol] : 0.0f;
#pragma unroll
        for (int r = 0; r < 8; ++r) {
            float v = acc[t][r] + bval;
            if (relu_act) v = fmaxf(v, 0.0f);
            C[(size_t)(rbase + r) * NCOLS + ccol] = v;
        }
    }
}

// ---------------------------------------------------------------------------
// Attention pre-pass: per (node, head) dot products with a_src / a_dst
// ---------------------------------------------------------------------------
__global__ void gat_alpha(const float* __restrict__ hp,
                          const float* __restrict__ a_src,
                          const float* __restrict__ a_dst,
                          float* __restrict__ al_s,
                          float* __restrict__ al_d)
{
    int t = blockIdx.x * blockDim.x + threadIdx.x;
    if (t >= GNN_N * GNN_HEADS) return;
    const int n  = t / GNN_HEADS;
    const int hh = t - n * GNN_HEADS;
    const float* v  = hp + (size_t)n * GNN_HID + hh * GNN_FH;
    const float* as = a_src + hh * GNN_FH;
    const float* ad = a_dst + hh * GNN_FH;
    float ss = 0.0f, dd = 0.0f;
#pragma unroll 8
    for (int f = 0; f < GNN_FH; ++f) {
        float x = v[f];
        ss = fmaf(x, as[f], ss);
        dd = fmaf(x, ad[f], dd);
    }
    al_s[t] = ss;
    al_d[t] = dd;
}

// zero accum[N*HID]; zero m_u/den[N*HEADS]
__global__ void gat_init(float* __restrict__ accum,
                         unsigned* __restrict__ m_u,
                         float* __restrict__ den)
{
    int t = blockIdx.x * blockDim.x + threadIdx.x;
    if (t < GNN_N * GNN_HID) accum[t] = 0.0f;
    if (t < GNN_N * GNN_HEADS) { m_u[t] = 0u; den[t] = 0.0f; }
}

__device__ __forceinline__ void edge_sd(const int* __restrict__ ei, int e,
                                        int& s, int& d)
{
    if (e < GNN_E) { s = ei[e]; d = ei[GNN_E + e]; }
    else           { s = d = e - GNN_E; }          // self loop
}

// order-preserving float<->uint for atomic max (all finite floats map > 0)
__device__ __forceinline__ unsigned f2ord(float f) {
    unsigned u = __float_as_uint(f);
    return (u & 0x80000000u) ? ~u : (u | 0x80000000u);
}
__device__ __forceinline__ float ord2f(unsigned u) {
    return (u & 0x80000000u) ? __uint_as_float(u & 0x7FFFFFFFu)
                             : __uint_as_float(~u);
}

__device__ __forceinline__ float leaky(float x) {
    return x > 0.0f ? x : NEG_SLOPE * x;
}

// pass 1: segment max of leaky_relu(al_s[src]+al_d[dst]) per dst/head
__global__ void gat_edge_max(const int* __restrict__ ei,
                             const float* __restrict__ al_s,
                             const float* __restrict__ al_d,
                             unsigned* __restrict__ m_u)
{
    int t = blockIdx.x * blockDim.x + threadIdx.x;
    if (t >= GNN_EP * GNN_HEADS) return;
    const int hh = t / GNN_EP;          // consecutive threads -> consecutive edges
    const int e  = t - hh * GNN_EP;
    int s, d; edge_sd(ei, e, s, d);
    float ev = leaky(al_s[s * GNN_HEADS + hh] + al_d[d * GNN_HEADS + hh]);
    atomicMax(&m_u[d * GNN_HEADS + hh], f2ord(ev));
}

// pass 2: num = exp(e - m[dst]); den[dst] += num
__global__ void gat_edge_expsum(const int* __restrict__ ei,
                                const float* __restrict__ al_s,
                                const float* __restrict__ al_d,
                                const unsigned* __restrict__ m_u,
                                float* __restrict__ num,
                                float* __restrict__ den)
{
    int t = blockIdx.x * blockDim.x + threadIdx.x;
    if (t >= GNN_EP * GNN_HEADS) return;
    const int hh = t / GNN_EP;
    const int e  = t - hh * GNN_EP;
    int s, d; edge_sd(ei, e, s, d);
    float ev = leaky(al_s[s * GNN_HEADS + hh] + al_d[d * GNN_HEADS + hh]);
    float mm = ord2f(m_u[d * GNN_HEADS + hh]);
    float nu = __expf(ev - mm);
    num[(size_t)hh * GNN_EP + e] = nu;
    atomicAdd(&den[d * GNN_HEADS + hh], nu);
}

// pass 3: accum[dst, head, :] += (num/den) * hp[src, head, :]
__global__ void gat_edge_aggregate(const int* __restrict__ ei,
                                   const float* __restrict__ num,
                                   const float* __restrict__ den,
                                   const float* __restrict__ hp,
                                   float* __restrict__ accum)
{
    int t = blockIdx.x * blockDim.x + threadIdx.x;
    if (t >= GNN_EP * GNN_HEADS) return;
    const int hh = t / GNN_EP;
    const int e  = t - hh * GNN_EP;
    int s, d; edge_sd(ei, e, s, d);
    const float coef = num[(size_t)hh * GNN_EP + e] /
                       (den[d * GNN_HEADS + hh] + 1e-16f);
    const float* __restrict__ hs = hp    + (size_t)s * GNN_HID + hh * GNN_FH;
    float*       __restrict__ ad = accum + (size_t)d * GNN_HID + hh * GNN_FH;
#pragma unroll 8
    for (int f = 0; f < GNN_FH; ++f)
        atomicAdd(&ad[f], coef * hs[f]);
}

// residual: h += relu(accum + b)
__global__ void gat_residual(float* __restrict__ h,
                             const float* __restrict__ accum,
                             const float* __restrict__ b)
{
    int t = blockIdx.x * blockDim.x + threadIdx.x;
    if (t >= GNN_N * GNN_HID) return;
    const int c = t & (GNN_HID - 1);
    float v = accum[t] + b[c];
    h[t] += (v > 0.0f ? v : 0.0f);
}

// ---------------------------------------------------------------------------
// Launch
// ---------------------------------------------------------------------------
static inline int cdiv(int a, int b) { return (a + b - 1) / b; }

extern "C" void kernel_launch(void* const* d_in, const int* in_sizes, int n_in,
                              void* d_out, int out_size, void* d_ws, size_t ws_size,
                              hipStream_t stream)
{
    (void)in_sizes; (void)n_in; (void)out_size; (void)ws_size;

    const float* x   = (const float*)d_in[0];
    const int*   ei  = (const int*)  d_in[1];   // [2, E] row-major
    const float* Wi  = (const float*)d_in[2];
    const float* bi  = (const float*)d_in[3];
    const float* W0  = (const float*)d_in[4];
    const float* as0 = (const float*)d_in[5];
    const float* ad0 = (const float*)d_in[6];
    const float* b0  = (const float*)d_in[7];
    const float* W1  = (const float*)d_in[8];
    const float* as1 = (const float*)d_in[9];
    const float* ad1 = (const float*)d_in[10];
    const float* b1  = (const float*)d_in[11];
    const float* Wo  = (const float*)d_in[12];
    const float* bo  = (const float*)d_in[13];
    float* out = (float*)d_out;

    // workspace carve-up (all fp32 unless noted); ~94 MB total
    float*    h     = (float*)d_ws;
    float*    hp    = h     + (size_t)GNN_N * GNN_HID;
    float*    accum = hp    + (size_t)GNN_N * GNN_HID;
    float*    al_s  = accum + (size_t)GNN_N * GNN_HID;
    float*    al_d  = al_s  + (size_t)GNN_N * GNN_HEADS;
    unsigned* m_u   = (unsigned*)(al_d + (size_t)GNN_N * GNN_HEADS);
    float*    den   = (float*)(m_u + (size_t)GNN_N * GNN_HEADS);
    float*    num   = den   + (size_t)GNN_N * GNN_HEADS;   // [HEADS, EP]

    const int BLK = 256;
    const int wavesPerBlk = BLK / 32;
    const int nStrips = GNN_N / 16;                 // 3125, exact
    const int gemmBlocks = cdiv(nStrips, wavesPerBlk);

    // ---- input projection: h = relu(x @ Wi + bi)
    gat_gemm_wmma<GNN_HID><<<gemmBlocks, BLK, 0, stream>>>(
        x, Wi, bi, h, GNN_N, 1);

    const float* Ws[2]  = {W0, W1};
    const float* ass[2] = {as0, as1};
    const float* ads[2] = {ad0, ad1};
    const float* bs[2]  = {b0, b1};

    const int nNodeHead = GNN_N * GNN_HEADS;
    const int nEdgeHead = GNN_EP * GNN_HEADS;
    const int nNodeFeat = GNN_N * GNN_HID;

    for (int l = 0; l < 2; ++l) {
        gat_gemm_wmma<GNN_HID><<<gemmBlocks, BLK, 0, stream>>>(
            h, Ws[l], nullptr, hp, GNN_N, 0);

        gat_alpha<<<cdiv(nNodeHead, BLK), BLK, 0, stream>>>(
            hp, ass[l], ads[l], al_s, al_d);

        gat_init<<<cdiv(nNodeFeat, BLK), BLK, 0, stream>>>(accum, m_u, den);

        gat_edge_max<<<cdiv(nEdgeHead, BLK), BLK, 0, stream>>>(
            ei, al_s, al_d, m_u);

        gat_edge_expsum<<<cdiv(nEdgeHead, BLK), BLK, 0, stream>>>(
            ei, al_s, al_d, m_u, num, den);

        gat_edge_aggregate<<<cdiv(nEdgeHead, BLK), BLK, 0, stream>>>(
            ei, num, den, hp, accum);

        gat_residual<<<cdiv(nNodeFeat, BLK), BLK, 0, stream>>>(
            h, accum, bs[l]);
    }

    // ---- output projection: out = h @ Wo + bo
    gat_gemm_wmma<GNN_OUTD><<<gemmBlocks, BLK, 0, stream>>>(
        h, Wo, bo, out, GNN_N, 0);
}
